// NysAttention_74680891343409
// MI455X (gfx1250) — compile-verified, matrix operand order checked by reading, not run on hardware
//
#include <hip/hip_runtime.h>
#include <hip/hip_bf16.h>
#include <math.h>

// ---------------------------------------------------------------------------
// Nystromformer attention on gfx1250 (MI455X), wave32 + WMMA.
//   B=4, N=4096, C=1024, H=16, hd=64, landmarks Lm=64, seg=64.
// Memory-bound problem (~320MB traffic vs ~11 GFLOP) -> bf16 WMMA GEMMs,
// fp32 V_WMMA_F32_16X16X4_F32 for the Newton-Schulz 64x64 inverse.
// ---------------------------------------------------------------------------

#define B_  4
#define N_  4096
#define C_  1024
#define H_  16
#define HD  64
#define LM  64
#define BH  64
#define CH  8      // N-chunks for the kernel_3 @ v streaming pass

typedef __attribute__((ext_vector_type(16))) __bf16 v16bf;
typedef __attribute__((ext_vector_type(8)))  float  v8f;
typedef __attribute__((ext_vector_type(2)))  float  v2f;

// ---- bf16 helpers ---------------------------------------------------------
__device__ __forceinline__ unsigned short f2bfbits(float f) {
  union { float f; unsigned u; } a; a.f = f;
  return (unsigned short)((a.u + 0x7fffu + ((a.u >> 16) & 1u)) >> 16);
}
__device__ __forceinline__ __bf16 bitsbf(unsigned short h) {
  return __builtin_bit_cast(__bf16, h);
}

// A/B fragment K-pair offsets within a 16-element (8-pair) fragment.
// pair j -> K offset: {0,2,4,6,16,18,20,22}  (==  j<4 ? 2j : 8+2j)
__device__ __forceinline__ int kOff(int j) { return (j < 4) ? (2 * j) : (8 + 2 * j); }

// Build a bf16 fragment from LDS (u16 array, K contiguous along last dim).
// Works for both A fragments (p = row base) and B fragments (p = col base of
// a K-major staging buffer); kbase = (lid>>4)*8 + kstep*32.
__device__ __forceinline__ v16bf frag_lds(const unsigned short* p, int kbase) {
  v16bf v;
#pragma unroll
  for (int j = 0; j < 8; ++j) {
    unsigned w = *(const unsigned*)(p + kbase + kOff(j));
    v[2 * j]     = bitsbf((unsigned short)(w & 0xffffu));
    v[2 * j + 1] = bitsbf((unsigned short)(w >> 16));
  }
  return v;
}

// Build a bf16 A fragment directly from global fp32 (row pointer), scaled.
__device__ __forceinline__ v16bf frag_gf32(const float* p, int kbase, float scale) {
  v16bf v;
#pragma unroll
  for (int j = 0; j < 8; ++j) {
    int o = kbase + kOff(j);
    v[2 * j]     = bitsbf(f2bfbits(p[o] * scale));
    v[2 * j + 1] = bitsbf(f2bfbits(p[o + 1] * scale));
  }
  return v;
}

__device__ __forceinline__ v8f wmma_bf(v16bf a, v16bf b, v8f c) {
  return __builtin_amdgcn_wmma_f32_16x16x32_bf16(false, a, false, b, (short)0, c,
                                                 false, false);
}

// ---- wave32 16-lane-group reductions (rows live across 16 lanes) ----------
__device__ __forceinline__ float redmax16(float v) {
#pragma unroll
  for (int m = 1; m < 16; m <<= 1) v = fmaxf(v, __shfl_xor(v, m, 32));
  return v;
}
__device__ __forceinline__ float redsum16(float v) {
#pragma unroll
  for (int m = 1; m < 16; m <<= 1) v += __shfl_xor(v, m, 32);
  return v;
}

// ---------------------------------------------------------------------------
// Kernel 1: landmark means.  qlm = mean(Q)/8 (pre-scaled), klm = mean(K).
// ---------------------------------------------------------------------------
__global__ __launch_bounds__(256) void k_landmarks(const float* __restrict__ Q,
                                                   const float* __restrict__ K,
                                                   float* __restrict__ qlm,
                                                   float* __restrict__ klm) {
  const int wg = blockIdx.x;          // bh*64 + l
  const int bh = wg >> 6, l = wg & 63;
  const int b = bh >> 4, h = bh & 15;
  const int t = threadIdx.x;
  const int d = t & 63, qtr = t >> 6;
  float sq = 0.f, sk = 0.f;
#pragma unroll 4
  for (int i = 0; i < 16; ++i) {
    int row = l * 64 + qtr * 16 + i;
    size_t g = ((size_t)b * N_ + row) * C_ + h * HD + d;
    sq += Q[g];
    sk += K[g];
  }
  __shared__ float redq[4][64];
  __shared__ float redk[4][64];
  redq[qtr][d] = sq;
  redk[qtr][d] = sk;
  __syncthreads();
  if (t < 64) {
    float a = redq[0][t] + redq[1][t] + redq[2][t] + redq[3][t];
    float c = redk[0][t] + redk[1][t] + redk[2][t] + redk[3][t];
    qlm[(size_t)bh * 4096 + l * 64 + t] = a * (1.0f / 64.0f) * 0.125f;
    klm[(size_t)bh * 4096 + l * 64 + t] = c * (1.0f / 64.0f);
  }
}

// ---------------------------------------------------------------------------
// fp32 64x64x64 matmul helper using V_WMMA_F32_16X16X4_F32 (16 k-steps).
// Wave wv owns output rows [16wv,16wv+16).  A,B are fp32 LDS, row-major.
// If btrans, B is stored [n][k] (i.e. B[k][n] read as Bm[n*64+k]).
// ---------------------------------------------------------------------------
__device__ __forceinline__ void mm64f32(v8f t[4], const float* A, const float* Bm,
                                        bool btrans, int wv, int lid) {
  const int m = wv * 16 + (lid & 15);
  const int kh = (lid >> 4) * 2;
  const int l15 = lid & 15;
  for (int ks = 0; ks < 16; ++ks) {
    const int k0 = ks * 4 + kh;
    v2f a;
    a.x = A[m * 64 + k0];
    a.y = A[m * 64 + k0 + 1];
#pragma unroll
    for (int c = 0; c < 4; ++c) {
      const int n = c * 16 + l15;
      v2f bb;
      if (btrans) { bb.x = Bm[n * 64 + k0];  bb.y = Bm[n * 64 + k0 + 1]; }
      else        { bb.x = Bm[k0 * 64 + n];  bb.y = Bm[(k0 + 1) * 64 + n]; }
      t[c] = __builtin_amdgcn_wmma_f32_16x16x4_f32(false, a, false, bb, (short)0,
                                                   t[c], false, false);
    }
  }
}

__device__ __forceinline__ void storeTiles(float* dst, const v8f t[4], int wv, int lid) {
  const int l15 = lid & 15, hi = lid >> 4;
#pragma unroll
  for (int c = 0; c < 4; ++c)
#pragma unroll
    for (int r = 0; r < 8; ++r)
      dst[(wv * 16 + r + 8 * hi) * 64 + c * 16 + l15] = t[c][r];
}
__device__ __forceinline__ void storeE(float* dst, const v8f t[4], float coef,
                                       int wv, int lid) {
  const int l15 = lid & 15, hi = lid >> 4;
#pragma unroll
  for (int c = 0; c < 4; ++c)
#pragma unroll
    for (int r = 0; r < 8; ++r) {
      int row = wv * 16 + r + 8 * hi, col = c * 16 + l15;
      dst[row * 64 + col] = (row == col ? coef : 0.0f) - t[c][r];
    }
}

// ---------------------------------------------------------------------------
// Kernel 2: kernel_2 = softmax(qlm@klm^T); Vinv = iterative_inv(kernel_2).
// One WG (4 waves) per (b,h).  All fp32 via v_wmma_f32_16x16x4_f32.
// ---------------------------------------------------------------------------
__global__ __launch_bounds__(128) void k_inv(const float* __restrict__ qlm,
                                             const float* __restrict__ klm,
                                             float* __restrict__ Vinv) {
  __shared__ float bK2[4096];
  __shared__ float bV[4096];
  __shared__ float bKV[4096];
  __shared__ float bE[4096];
  const int bh = blockIdx.x;
  const int tid = threadIdx.x, wv = tid >> 5, lid = tid & 31;
  const int l15 = lid & 15, hi = lid >> 4;

  const float* qsrc = qlm + (size_t)bh * 4096;
  const float* ksrc = klm + (size_t)bh * 4096;
  for (int i = tid; i < 4096; i += 128) { bV[i] = qsrc[i]; bKV[i] = ksrc[i]; }
  __syncthreads();

  const v8f vzero = {};
  v8f t[4];

  // S = qlm @ klm^T   (B[k][n] = klm[n][k] -> btrans on row-major klm)
#pragma unroll
  for (int c = 0; c < 4; ++c) t[c] = vzero;
  mm64f32(t, bV, bKV, true, wv, lid);

  // row softmax -> bK2
  {
    float mr[8];
#pragma unroll
    for (int r = 0; r < 8; ++r)
      mr[r] = redmax16(fmaxf(fmaxf(t[0][r], t[1][r]), fmaxf(t[2][r], t[3][r])));
#pragma unroll
    for (int r = 0; r < 8; ++r) {
      float s0 = 0.f;
#pragma unroll
      for (int c = 0; c < 4; ++c) { float p = __expf(t[c][r] - mr[r]); t[c][r] = p; s0 += p; }
      float inv = 1.0f / redsum16(s0);
#pragma unroll
      for (int c = 0; c < 4; ++c) t[c][r] *= inv;
    }
    storeTiles(bK2, t, wv, lid);
  }
  __syncthreads();

  // denom = max_j colsum(K2);  Vm = K2^T / denom
  if (tid < 64) {
    float cs = 0.f;
    for (int i = 0; i < 64; ++i) cs += bK2[i * 64 + tid];
    bE[tid] = cs;
  }
  __syncthreads();
  float dmax = 0.f;
  for (int j = 0; j < 64; ++j) dmax = fmaxf(dmax, bE[j]);
  const float dinv = 1.0f / dmax;
  __syncthreads();
  for (int i = tid; i < 4096; i += 128) {
    int rI = i >> 6, cJ = i & 63;
    bV[i] = bK2[cJ * 64 + rI] * dinv;
  }
  __syncthreads();

  // 6 Newton-Schulz iterations: Vm = 0.25*Vm@(13I - KV@(15I - KV@(7I - KV)))
  for (int it = 0; it < 6; ++it) {
#pragma unroll
    for (int c = 0; c < 4; ++c) t[c] = vzero;
    mm64f32(t, bK2, bV, false, wv, lid);           // KV = K2 @ Vm
    __syncthreads();
    storeTiles(bKV, t, wv, lid);
    storeE(bE, t, 7.0f, wv, lid);                  // E1 = 7I - KV
    __syncthreads();
#pragma unroll
    for (int c = 0; c < 4; ++c) t[c] = vzero;
    mm64f32(t, bKV, bE, false, wv, lid);           // M1 = KV @ E1
    __syncthreads();
    storeE(bE, t, 15.0f, wv, lid);                 // E2 = 15I - M1
    __syncthreads();
#pragma unroll
    for (int c = 0; c < 4; ++c) t[c] = vzero;
    mm64f32(t, bKV, bE, false, wv, lid);           // M2 = KV @ E2
    __syncthreads();
    storeE(bE, t, 13.0f, wv, lid);                 // E3 = 13I - M2
    __syncthreads();
#pragma unroll
    for (int c = 0; c < 4; ++c) t[c] = vzero;
    mm64f32(t, bV, bE, false, wv, lid);            // Vm @ E3
    __syncthreads();
#pragma unroll
    for (int c = 0; c < 4; ++c)
#pragma unroll
      for (int r = 0; r < 8; ++r) t[c][r] *= 0.25f;
    storeTiles(bV, t, wv, lid);
    __syncthreads();
  }

  float* dst = Vinv + (size_t)bh * 4096;
  for (int i = tid; i < 4096; i += 128) dst[i] = bV[i];
}

// ---------------------------------------------------------------------------
// Kernel 3: chunked streaming  softmax(qlm @ k^T) @ v  with online softmax.
// Grid = BH*CH; chunk ch handles 512 keys (8 blocks of 64). bf16 WMMA.
// Emits unnormalized partial acc + per-row (m, s) stats.
// ---------------------------------------------------------------------------
__global__ __launch_bounds__(128) void k_kv_partial(const float* __restrict__ K,
                                                    const float* __restrict__ V,
                                                    const float* __restrict__ qlm,
                                                    float* __restrict__ pacc,
                                                    float* __restrict__ pstat) {
  const int blk = blockIdx.x;
  const int bh = blk / CH, ch = blk % CH;
  const int b = bh >> 4, h = bh & 15;
  const int tid = threadIdx.x, wv = tid >> 5, lid = tid & 31;
  const int l15 = lid & 15, hi = lid >> 4, kb = hi * 8;

  __shared__ alignas(16) unsigned short k_s[4096];       // [n][d] bf16
  __shared__ alignas(16) unsigned short v_t[4096];       // [d][n] bf16 (transposed)
  __shared__ alignas(16) unsigned short P_s[4][1024];    // per-wave P staging

  // qlm A fragments (already scaled by 1/8)
  v16bf aq[2];
  {
    const float* qp = qlm + (size_t)bh * 4096 + (size_t)(wv * 16 + l15) * 64;
#pragma unroll
    for (int ks = 0; ks < 2; ++ks) aq[ks] = frag_gf32(qp, ks * 32 + kb, 1.0f);
  }

  const v8f vzero = {};
  float mrow[8], srow[8], scl[8];
  v8f acc[4];
#pragma unroll
  for (int r = 0; r < 8; ++r) { mrow[r] = -1e30f; srow[r] = 0.f; }
#pragma unroll
  for (int c = 0; c < 4; ++c) acc[c] = vzero;

  for (int nb = ch * 8; nb < ch * 8 + 8; ++nb) {
    __syncthreads();
    for (int i = tid; i < 4096; i += 128) {
      int n = i >> 6, d = i & 63;
      size_t g = ((size_t)b * N_ + (size_t)nb * 64 + n) * C_ + h * HD + d;
      k_s[n * 64 + d] = f2bfbits(K[g]);
      v_t[d * 64 + n] = f2bfbits(V[g]);
    }
    __syncthreads();

    v8f s[4];
#pragma unroll
    for (int c = 0; c < 4; ++c) {
      s[c] = vzero;
      const unsigned short* kr = k_s + (c * 16 + l15) * 64;
#pragma unroll
      for (int ks = 0; ks < 2; ++ks)
        s[c] = wmma_bf(aq[ks], frag_lds(kr, ks * 32 + kb), s[c]);
    }

    // online softmax update
#pragma unroll
    for (int r = 0; r < 8; ++r) {
      float mx = fmaxf(fmaxf(s[0][r], s[1][r]), fmaxf(s[2][r], s[3][r]));
      mx = redmax16(mx);
      float mnew = fmaxf(mrow[r], mx);
      scl[r] = __expf(mrow[r] - mnew);
      mrow[r] = mnew;
      srow[r] *= scl[r];
    }
#pragma unroll
    for (int r = 0; r < 8; ++r) {
      float s0 = 0.f;
#pragma unroll
      for (int c = 0; c < 4; ++c) { float p = __expf(s[c][r] - mrow[r]); s[c][r] = p; s0 += p; }
      srow[r] += redsum16(s0);
    }
#pragma unroll
    for (int c = 0; c < 4; ++c)
#pragma unroll
      for (int r = 0; r < 8; ++r)
        P_s[wv][(r + 8 * hi) * 64 + c * 16 + l15] = f2bfbits(s[c][r]);
    __syncthreads();

    v16bf ap[2];
    {
      const unsigned short* pr = &P_s[wv][l15 * 64];
#pragma unroll
      for (int ks = 0; ks < 2; ++ks) ap[ks] = frag_lds(pr, ks * 32 + kb);
    }
#pragma unroll
    for (int c = 0; c < 4; ++c) {
#pragma unroll
      for (int r = 0; r < 8; ++r) acc[c][r] *= scl[r];
      const unsigned short* vr = v_t + (c * 16 + l15) * 64;
#pragma unroll
      for (int ks = 0; ks < 2; ++ks)
        acc[c] = wmma_bf(ap[ks], frag_lds(vr, ks * 32 + kb), acc[c]);
    }
  }

  float* pa = pacc + (size_t)blk * 4096;
#pragma unroll
  for (int c = 0; c < 4; ++c)
#pragma unroll
    for (int r = 0; r < 8; ++r)
      pa[(wv * 16 + r + 8 * hi) * 64 + c * 16 + l15] = acc[c][r];
  if (l15 == 0) {
    float* st = pstat + (size_t)blk * 128;
#pragma unroll
    for (int r = 0; r < 8; ++r) {
      int row = wv * 16 + r + 8 * hi;
      st[row] = mrow[r];
      st[64 + row] = srow[r];
    }
  }
}

// ---------------------------------------------------------------------------
// Kernel 3b: log-sum-exp merge of the CH chunk partials -> KV (normalized).
// ---------------------------------------------------------------------------
__global__ __launch_bounds__(128) void k_kv_merge(const float* __restrict__ pacc,
                                                  const float* __restrict__ pstat,
                                                  float* __restrict__ KVout) {
  const int bh = blockIdx.x;
  const int tid = threadIdx.x;
  __shared__ float w[CH][64];
  __shared__ float invd[64];
  if (tid < 64) {
    const int l = tid;
    float m = -1e30f;
    for (int c = 0; c < CH; ++c)
      m = fmaxf(m, pstat[(size_t)(bh * CH + c) * 128 + l]);
    float den = 0.f;
    for (int c = 0; c < CH; ++c) {
      float e = __expf(pstat[(size_t)(bh * CH + c) * 128 + l] - m);
      w[c][l] = e;
      den += pstat[(size_t)(bh * CH + c) * 128 + 64 + l] * e;
    }
    invd[l] = 1.0f / den;
  }
  __syncthreads();
  for (int i = tid; i < 4096; i += 128) {
    const int l = i >> 6;
    float s = 0.f;
    for (int c = 0; c < CH; ++c)
      s += pacc[(size_t)(bh * CH + c) * 4096 + i] * w[c][l];
    KVout[(size_t)bh * 4096 + i] = s * invd[l];
  }
}

// ---------------------------------------------------------------------------
// Kernel 4: out = softmax(q@klm^T) @ Vinv @ KV.  64 q-rows per WG, bf16 WMMA.
// ---------------------------------------------------------------------------
__global__ __launch_bounds__(128) void k_out(const float* __restrict__ Q,
                                             const float* __restrict__ klm,
                                             const float* __restrict__ Vinv,
                                             const float* __restrict__ KVin,
                                             float* __restrict__ out) {
  const int wg = blockIdx.x;
  const int bh = wg >> 6, nblk = wg & 63;
  const int b = bh >> 4, h = bh & 15;
  const int tid = threadIdx.x, wv = tid >> 5, lid = tid & 31;
  const int l15 = lid & 15, hi = lid >> 4, kb = hi * 8;

  __shared__ alignas(16) unsigned short klm_s[4096];    // [lm][d]
  __shared__ alignas(16) unsigned short vinv_t[4096];   // [n][k]  (Vinv^T)
  __shared__ alignas(16) unsigned short kv_t[4096];     // [d][l]  (KV^T)
  __shared__ alignas(16) unsigned short P_s[4][1024];

  for (int i = tid; i < 4096; i += 128) {
    int r0 = i >> 6, c0 = i & 63;
    klm_s[i]            = f2bfbits(klm [(size_t)bh * 4096 + i]);
    vinv_t[c0 * 64 + r0] = f2bfbits(Vinv[(size_t)bh * 4096 + i]);
    kv_t  [c0 * 64 + r0] = f2bfbits(KVin[(size_t)bh * 4096 + i]);
  }

  // q A fragments, scaled by 1/sqrt(hd) = 1/8
  v16bf aq[2];
  {
    const int row = nblk * 64 + wv * 16 + l15;
    const float* qp = Q + ((size_t)b * N_ + row) * C_ + h * HD;
#pragma unroll
    for (int ks = 0; ks < 2; ++ks) aq[ks] = frag_gf32(qp, ks * 32 + kb, 0.125f);
  }
  __syncthreads();

  const v8f vzero = {};
  // S = q @ klm^T
  v8f s[4];
#pragma unroll
  for (int c = 0; c < 4; ++c) {
    s[c] = vzero;
    const unsigned short* kr = klm_s + (c * 16 + l15) * 64;
#pragma unroll
    for (int ks = 0; ks < 2; ++ks)
      s[c] = wmma_bf(aq[ks], frag_lds(kr, ks * 32 + kb), s[c]);
  }
  // full-row softmax (row length 64)
  float mr[8];
#pragma unroll
  for (int r = 0; r < 8; ++r)
    mr[r] = redmax16(fmaxf(fmaxf(s[0][r], s[1][r]), fmaxf(s[2][r], s[3][r])));
#pragma unroll
  for (int r = 0; r < 8; ++r) {
    float s0 = 0.f;
#pragma unroll
    for (int c = 0; c < 4; ++c) { float p = __expf(s[c][r] - mr[r]); s[c][r] = p; s0 += p; }
    float inv = 1.0f / redsum16(s0);
#pragma unroll
    for (int c = 0; c < 4; ++c) s[c][r] *= inv;
  }
#pragma unroll
  for (int c = 0; c < 4; ++c)
#pragma unroll
    for (int r = 0; r < 8; ++r)
      P_s[wv][(r + 8 * hi) * 64 + c * 16 + l15] = f2bfbits(s[c][r]);
  __syncthreads();

  // T = P @ Vinv
  v16bf ap[2];
  {
    const unsigned short* pr = &P_s[wv][l15 * 64];
#pragma unroll
    for (int ks = 0; ks < 2; ++ks) ap[ks] = frag_lds(pr, ks * 32 + kb);
  }
  v8f tt[4];
#pragma unroll
  for (int c = 0; c < 4; ++c) {
    tt[c] = vzero;
    const unsigned short* br = vinv_t + (c * 16 + l15) * 64;
#pragma unroll
    for (int ks = 0; ks < 2; ++ks)
      tt[c] = wmma_bf(ap[ks], frag_lds(br, ks * 32 + kb), tt[c]);
  }
  __syncthreads();
#pragma unroll
  for (int c = 0; c < 4; ++c)
#pragma unroll
    for (int r = 0; r < 8; ++r)
      P_s[wv][(r + 8 * hi) * 64 + c * 16 + l15] = f2bfbits(tt[c][r]);
  __syncthreads();

  // O = T @ KV
  v16bf at[2];
  {
    const unsigned short* pr = &P_s[wv][l15 * 64];
#pragma unroll
    for (int ks = 0; ks < 2; ++ks) at[ks] = frag_lds(pr, ks * 32 + kb);
  }
  v8f o[4];
#pragma unroll
  for (int c = 0; c < 4; ++c) {
    o[c] = vzero;
    const unsigned short* br = kv_t + (c * 16 + l15) * 64;
#pragma unroll
    for (int ks = 0; ks < 2; ++ks)
      o[c] = wmma_bf(at[ks], frag_lds(br, ks * 32 + kb), o[c]);
  }
#pragma unroll
  for (int c = 0; c < 4; ++c)
#pragma unroll
    for (int r = 0; r < 8; ++r) {
      int row = nblk * 64 + wv * 16 + r + 8 * hi;
      int d = c * 16 + l15;
      out[((size_t)b * N_ + row) * C_ + h * HD + d] = o[c][r];
    }
}

// ---------------------------------------------------------------------------
extern "C" void kernel_launch(void* const* d_in, const int* in_sizes, int n_in,
                              void* d_out, int out_size, void* d_ws, size_t ws_size,
                              hipStream_t stream) {
  (void)in_sizes; (void)n_in; (void)out_size; (void)ws_size;
  const float* Q = (const float*)d_in[0];
  const float* K = (const float*)d_in[1];
  const float* V = (const float*)d_in[2];
  float* out = (float*)d_out;
  float* ws  = (float*)d_ws;

  // fp32 workspace layout (floats):
  float* qlm   = ws;                         // 64*64*64
  float* klm   = ws + 262144;                // 64*64*64
  float* Vinv  = ws + 524288;                // 64*64*64
  float* KVw   = ws + 786432;                // 64*64*64
  float* pacc  = ws + 1048576;               // 512*4096
  float* pstat = ws + 1048576 + 512 * 4096;  // 512*128

  k_landmarks<<<dim3(BH * LM), dim3(256), 0, stream>>>(Q, K, qlm, klm);
  k_inv      <<<dim3(BH),      dim3(128), 0, stream>>>(qlm, klm, Vinv);
  k_kv_partial<<<dim3(BH * CH), dim3(128), 0, stream>>>(K, V, qlm, pacc, pstat);
  k_kv_merge <<<dim3(BH),      dim3(128), 0, stream>>>(pacc, pstat, KVw);
  k_out      <<<dim3(BH * 64), dim3(128), 0, stream>>>(Q, klm, Vinv, KVw, out);
}
